// RotaryAttention_19816979104475
// MI455X (gfx1250) — matrix operand order repaired and checked by reading, not compiled, not run on hardware
//
#include <hip/hip_runtime.h>

// ---------------- problem constants ----------------
#define B    2
#define S    2048
#define H    16
#define D    128
#define R    16
#define HID  (H * D)
#define KVP  (S + 32)            // kv rows padded to a multiple of 32
#define SCALE 0.08838834764831845f   // 1/sqrt(128)

// LDS row strides (halfs) — non-power-of-two to avoid bank conflicts
#define KSTR 136                 // 32 x 136 halfs
#define VSTR 40                  // 128 x 40 halfs
#define PSTR 40                  // 4 x 16 x 40

// workspace layout (bytes)
#define QH_BYTES ((size_t)B * H * S   * D * 2)   // 16,777,216
#define KH_BYTES ((size_t)B * H * KVP * D * 2)   // 17,039,360

typedef __attribute__((ext_vector_type(16))) _Float16 v16h;
typedef __attribute__((ext_vector_type(8)))  _Float16 v8h;
typedef __attribute__((ext_vector_type(8)))  float    v8f;

__device__ inline v16h cat8(v8h lo, v8h hi) {
    return __builtin_shufflevector(lo, hi, 0,1,2,3,4,5,6,7,8,9,10,11,12,13,14,15);
}

// 16 contiguous halfs, 16B-aligned (padded rows are not 32B-aligned)
__device__ inline v16h ldfrag(const _Float16* p) {
    v8h lo = *(const v8h*)p;
    v8h hi = *(const v8h*)(p + 8);
    return cat8(lo, hi);
}

__device__ inline v8f wmma16(v16h a, v16h b, v8f c) {
    return __builtin_amdgcn_wmma_f32_16x16x32_f16(
        /*neg_a=*/false, a, /*neg_b=*/false, b,
        /*c_mod=*/(short)0, c, /*reuse_a=*/false, /*reuse_b=*/false);
}

// async global -> LDS DMA, 16B per lane (ASYNCcnt-tracked; cdna5 §08.4)
__device__ inline void async_b128(_Float16* lds_dst, const _Float16* gsrc) {
    unsigned loff = (unsigned)(unsigned long long)lds_dst;
    asm volatile("global_load_async_to_lds_b128 %0, %1, off"
                 :: "v"(loff), "v"(gsrc)
                 : "memory");
}

__device__ inline void wait_async0() {
    asm volatile("s_wait_asynccnt 0x0" ::: "memory");
}

// ---------------------------------------------------------------------------
// Kernel 1: RoPE on q,k + f32->f16 pack into (B,H,rows,D) layout
// ---------------------------------------------------------------------------
__global__ __launch_bounds__(256)
void rope_pack_kernel(const float* __restrict__ q, const float* __restrict__ k,
                      const int* __restrict__ pos,
                      _Float16* __restrict__ qh, _Float16* __restrict__ kh)
{
    size_t t = (size_t)blockIdx.x * 256 + threadIdx.x;   // B*H*S*D threads
    int d = (int)(t & (D - 1));
    int s = (int)((t >> 7) & (S - 1));
    int h = (int)((t >> 18) & (H - 1));
    int b = (int)(t >> 22);

    float p = (float)pos[b * S + s];
    int   i = d & 63;
    float ang = p * __powf(10000.0f, -(float)(2 * i) * (1.0f / 128.0f));
    float sn, cs;
    __sincosf(ang, &sn, &cs);

    size_t base = ((size_t)(b * S + s)) * HID + h * D;
    float qv = q[base + d], qp = q[base + (d ^ 64)];
    float kv = k[base + d], kp = k[base + (d ^ 64)];
    float qrot = (d < 64) ? -qp : qp;
    float krot = (d < 64) ? -kp : kp;

    size_t qo = (((size_t)(b * H + h)) * S   + s) * D + d;
    size_t ko = (((size_t)(b * H + h)) * KVP + s) * D + d;
    qh[qo] = (_Float16)(qv * cs + qrot * sn);
    kh[ko] = (_Float16)(kv * cs + krot * sn);
}

// ---------------------------------------------------------------------------
// Kernel 2: LDS-tiled transpose+convert: v (B,S,H*D) f32 -> vt (B,H,D,KVP) f16
// ---------------------------------------------------------------------------
__global__ __launch_bounds__(128)
void v_transpose_kernel(const float* __restrict__ v, _Float16* __restrict__ vt)
{
    __shared__ alignas(16) _Float16 tile[32 * 136];      // [s][d], padded rows

    const int sblk = blockIdx.x;
    const int bh   = blockIdx.y;
    const int b = bh >> 4, h = bh & 15;
    const int tid = threadIdx.x;
    const int s0 = sblk * 32;

#pragma unroll
    for (int it = 0; it < 4; ++it) {
        int idx = it * 128 + tid;
        int r   = idx >> 4;
        int dc  = (idx & 15) * 8;
        const float* src = v + ((size_t)(b * S + s0 + r)) * HID + h * D + dc;
        v8h tmp;
#pragma unroll
        for (int j = 0; j < 8; ++j) tmp[j] = (_Float16)src[j];
        *(v8h*)(&tile[r * 136 + dc]) = tmp;
    }
    __syncthreads();
    {
        int d = tid;
        v8h o0, o1, o2, o3;
#pragma unroll
        for (int j = 0; j < 8; ++j) {
            o0[j] = tile[(j)      * 136 + d];
            o1[j] = tile[(8 + j)  * 136 + d];
            o2[j] = tile[(16 + j) * 136 + d];
            o3[j] = tile[(24 + j) * 136 + d];
        }
        _Float16* dst = vt + (((size_t)bh) * D + d) * KVP + s0;
        *(v8h*)(dst)      = o0;
        *(v8h*)(dst + 8)  = o1;
        *(v8h*)(dst + 16) = o2;
        *(v8h*)(dst + 24) = o3;
    }
}

// ---------------------------------------------------------------------------
// Kernel 3: register rows (no RoPE) + zero pad rows [S+R, KVP)
// ---------------------------------------------------------------------------
__global__ __launch_bounds__(256)
void pack_regs_kernel(const float* __restrict__ kreg, const float* __restrict__ vreg,
                      _Float16* __restrict__ kh, _Float16* __restrict__ vt)
{
    int t = blockIdx.x * 256 + threadIdx.x;
    int d = t & 127;
    int r = (t >> 7) & 31;
    int h = (t >> 12) & 15;
    int b = t >> 16;

    float kv = 0.0f, vv = 0.0f;
    if (r < R) {
        kv = kreg[(h * R + r) * D + d];
        vv = vreg[(h * R + r) * D + d];
    }
    int bh = b * H + h;
    kh[(((size_t)bh) * KVP + S + r) * D + d]   = (_Float16)kv;
    vt[(((size_t)bh) * D + d) * KVP + (S + r)] = (_Float16)vv;
}

// ---------------------------------------------------------------------------
// async tile staging: K (32x128) + V^T (128x32), 16B per lane per issue
// ---------------------------------------------------------------------------
__device__ inline void issue_tile_async(const _Float16* kbase, const _Float16* vtbase,
                                        int key0, int tid,
                                        _Float16* Kb, _Float16* Vb)
{
#pragma unroll
    for (int rr = 0; rr < 4; ++rr) {
        int idx  = rr * 128 + tid;
        int krow = idx >> 4;
        int kcol = (idx & 15) * 8;
        async_b128(&Kb[krow * KSTR + kcol],
                   kbase + (size_t)(key0 + krow) * D + kcol);
    }
#pragma unroll
    for (int rr = 0; rr < 4; ++rr) {
        int idx  = rr * 128 + tid;
        int drow = idx >> 2;
        int koff = (idx & 3) * 8;
        async_b128(&Vb[drow * VSTR + koff],
                   vtbase + (size_t)drow * KVP + key0 + koff);
    }
}

// ---------------------------------------------------------------------------
// Kernel 4: flash attention, double-buffered async tiles, 4 waves / block
// ---------------------------------------------------------------------------
__global__ __launch_bounds__(128)
void fa_kernel(const _Float16* __restrict__ qh, const _Float16* __restrict__ kh,
               const _Float16* __restrict__ vt, float* __restrict__ out)
{
    __shared__ alignas(32) _Float16 ldsK [2][32  * KSTR];   // 2 x 8704 B
    __shared__ alignas(32) _Float16 ldsVt[2][128 * VSTR];   // 2 x 10240 B
    __shared__ alignas(32) _Float16 ldsP [4 * 16 * PSTR];   // 5120 B

    const int qblk = blockIdx.x;
    const int bh   = blockIdx.y;
    const int bb   = bh >> 4;
    const int h    = bh & 15;
    const int tid  = threadIdx.x;
    const int wave = tid >> 5;
    const int lane = tid & 31;
    const int l16  = lane & 15;
    const int hi16 = lane >> 4;

    const int q0 = qblk * 64 + wave * 16;

    const _Float16* qbase  = qh + (((size_t)bh) * S + q0) * D;
    const _Float16* kbase  = kh + ((size_t)bh) * KVP * D;
    const _Float16* vtbase = vt + ((size_t)bh) * D * KVP;

    // ---- Q A-fragments (16x128 -> 4 chunks of 16x32), kept in VGPRs ----
    v16h qfrag[4];
#pragma unroll
    for (int c = 0; c < 4; ++c) {
        int c0 = c * 32 + hi16 * 16;
        v8h lo = *(const v8h*)(qbase + (size_t)l16 * D + c0);
        v8h hi = *(const v8h*)(qbase + (size_t)l16 * D + c0 + 8);
        qfrag[c] = cat8(lo, hi);
    }

    v8f o[8];
    float m_r[8], l_r[8];
#pragma unroll
    for (int n = 0; n < 8; ++n) o[n] = (v8f){0,0,0,0,0,0,0,0};
#pragma unroll
    for (int i = 0; i < 8; ++i) { m_r[i] = -3.0e38f; l_r[i] = 0.0f; }

    const int nkb = qblk * 2 + 2;

    // ---- prologue: async-stage tile 0 into buffer 0 ----
    issue_tile_async(kbase, vtbase, 0, tid, &ldsK[0][0], &ldsVt[0][0]);

    for (int it = 0; it <= nkb; ++it) {
        const int buf  = it & 1;
        const int kb   = (it == nkb) ? (S / 32) : it;
        const int key0 = kb * 32;

        wait_async0();                         // my async copies for `buf` done
        __syncthreads();                       // -> whole tile visible

        // async-stage next tile into the other buffer (overlaps compute)
        if (it < nkb) {
            int kbn = (it + 1 == nkb) ? (S / 32) : (it + 1);
            issue_tile_async(kbase, vtbase, kbn * 32, tid,
                             &ldsK[buf ^ 1][0], &ldsVt[buf ^ 1][0]);
        }

        const _Float16* Kb = &ldsK[buf][0];
        const _Float16* Vb = &ldsVt[buf][0];

        // wave-uniform specialization
        const bool active = (key0 <= q0 + 15) || (key0 >= S);
        const bool nomask = (key0 + 31 <= q0);

        if (active) {
            // ---- scores: software-pipelined frag loads vs WMMAs ----
            v8f s0 = (v8f){0,0,0,0,0,0,0,0};
            v8f s1 = (v8f){0,0,0,0,0,0,0,0};
            const _Float16* kr0 = &Kb[l16 * KSTR        + hi16 * 16];
            const _Float16* kr1 = &Kb[(16 + l16) * KSTR + hi16 * 16];
            v16h bk0 = ldfrag(kr0);
            v16h bk1 = ldfrag(kr1);
#pragma unroll
            for (int c = 0; c < 4; ++c) {
                v16h nx0, nx1;
                if (c < 3) {                   // issue next chunk before use
                    nx0 = ldfrag(kr0 + (c + 1) * 32);
                    nx1 = ldfrag(kr1 + (c + 1) * 32);
                }
                s0 = wmma16(qfrag[c], bk0, s0);
                s1 = wmma16(qfrag[c], bk1, s1);
                if (c < 3) { bk0 = nx0; bk1 = nx1; }
            }

            // ---- mask (only when needed) + online softmax ----
            float sa[8], sb[8], ratio[8];
            if (nomask) {
#pragma unroll
                for (int i = 0; i < 8; ++i) {
                    sa[i] = s0[i] * SCALE;
                    sb[i] = s1[i] * SCALE;
                }
            } else {
                const int ka = key0 + l16;
                const int kc = key0 + 16 + l16;
#pragma unroll
                for (int i = 0; i < 8; ++i) {
                    int row = q0 + i + hi16 * 8;
                    float x = s0[i] * SCALE;
                    float y = s1[i] * SCALE;
                    if (!((ka <= row) || (ka >= S && ka < S + R))) x += -1.0e9f;
                    if (!((kc <= row) || (kc >= S && kc < S + R))) y += -1.0e9f;
                    sa[i] = x; sb[i] = y;
                }
            }
#pragma unroll
            for (int i = 0; i < 8; ++i) {
                float t = fmaxf(sa[i], sb[i]);
                t = fmaxf(t, __shfl_xor(t, 1));
                t = fmaxf(t, __shfl_xor(t, 2));
                t = fmaxf(t, __shfl_xor(t, 4));
                t = fmaxf(t, __shfl_xor(t, 8));
                float mn = fmaxf(m_r[i], t);
                ratio[i] = __expf(m_r[i] - mn);
                m_r[i]   = mn;
                float pa = __expf(sa[i] - mn);
                float pb = __expf(sb[i] - mn);
                float rs = pa + pb;
                rs += __shfl_xor(rs, 1);
                rs += __shfl_xor(rs, 2);
                rs += __shfl_xor(rs, 4);
                rs += __shfl_xor(rs, 8);
                l_r[i] = l_r[i] * ratio[i] + rs;
                ldsP[wave * (16 * PSTR) + (i + hi16 * 8) * PSTR + l16]      = (_Float16)pa;
                ldsP[wave * (16 * PSTR) + (i + hi16 * 8) * PSTR + 16 + l16] = (_Float16)pb;
            }
#pragma unroll
            for (int n = 0; n < 8; ++n)
#pragma unroll
                for (int i = 0; i < 8; ++i) o[n][i] *= ratio[i];
        }

        __syncthreads();                       // P visible (uniform barrier)

        if (active) {
            // ---- P x V: software-pipelined frag loads vs WMMAs ----
            v8h plo = *(const v8h*)(&ldsP[wave * (16 * PSTR) + l16 * PSTR + hi16 * 16]);
            v8h phi = *(const v8h*)(&ldsP[wave * (16 * PSTR) + l16 * PSTR + hi16 * 16 + 8]);
            v16h pfrag = cat8(plo, phi);
            const _Float16* vr = &Vb[l16 * VSTR + hi16 * 16];
            v16h bv = ldfrag(vr);
#pragma unroll
            for (int n = 0; n < 8; ++n) {
                v16h nx;
                if (n < 7) nx = ldfrag(vr + (n + 1) * (16 * VSTR));
                o[n] = wmma16(pfrag, bv, o[n]);
                if (n < 7) bv = nx;
            }
        }
    }

    // ---- epilogue: normalize and store (B,S,H*D) f32 ----
    float* obase = out + ((size_t)bb) * S * HID + h * D;
#pragma unroll
    for (int n = 0; n < 8; ++n)
#pragma unroll
        for (int i = 0; i < 8; ++i) {
            int row = q0 + i + hi16 * 8;
            obase[(size_t)row * HID + n * 16 + l16] = o[n][i] / l_r[i];
        }
}

// ---------------------------------------------------------------------------
extern "C" void kernel_launch(void* const* d_in, const int* in_sizes, int n_in,
                              void* d_out, int out_size, void* d_ws, size_t ws_size,
                              hipStream_t stream)
{
    const float* q    = (const float*)d_in[0];
    const float* k    = (const float*)d_in[1];
    const float* v    = (const float*)d_in[2];
    const int*   pos  = (const int*)  d_in[3];
    // d_in[4] = attention_mask: analytically causal, never read
    const float* kreg = (const float*)d_in[5];
    const float* vreg = (const float*)d_in[6];
    float* out = (float*)d_out;

    char* ws = (char*)d_ws;
    _Float16* qh = (_Float16*)(ws);
    _Float16* kh = (_Float16*)(ws + QH_BYTES);
    _Float16* vt = (_Float16*)(ws + QH_BYTES + KH_BYTES);

    rope_pack_kernel<<<(B * H * S * D) / 256, 256, 0, stream>>>(q, k, pos, qh, kh);

    dim3 tgrid(S / 32, B * H);
    v_transpose_kernel<<<tgrid, 128, 0, stream>>>(v, vt);

    pack_regs_kernel<<<(B * H * 32 * D) / 256, 256, 0, stream>>>(kreg, vreg, kh, vt);

    dim3 grid(S / 64, B * H);
    fa_kernel<<<grid, 128, 0, stream>>>(qh, kh, vt, out);
}
